// Conv2dQInt8Module_28166395527285
// MI455X (gfx1250) — 8-bit matrix operand order repaired and checked by reading, compiled, not run on hardware
//
#include <hip/hip_runtime.h>
#include <cstdint>
#include <cstddef>


typedef __attribute__((ext_vector_type(8))) int v8i;
typedef __attribute__((ext_vector_type(4))) int v4i;

#define N_BATCH   32
#define C_IN      128
#define H_IN      56
#define W_IN      56
#define C_OUT     256
#define H_OUT     54
#define W_OUT     54
#define K_TOTAL   1152
#define N_CHUNKS  18
#define PIXELS    (N_BATCH * H_OUT * W_OUT)
#define PIX_TILES (PIXELS / 16)
#define NHWC_BYTES ((size_t)N_BATCH * H_IN * W_IN * C_IN)
#define WFRAG_BYTES ((size_t)C_OUT * K_TOTAL)

__global__ void __launch_bounds__(128) repack_input_kernel(
    const int8_t* __restrict__ in, int8_t* __restrict__ nhwc) {
  const int c   = threadIdx.x;
  const int pix = blockIdx.x;
  const int n   = pix / (H_IN * W_IN);
  const int hw  = pix - n * (H_IN * W_IN);
  nhwc[(size_t)pix * C_IN + c] =
      in[((size_t)n * C_IN + c) * (H_IN * W_IN) + hw];
}

__global__ void __launch_bounds__(256) repack_weight_frag_kernel(
    const int8_t* __restrict__ w, int8_t* __restrict__ wfrag) {
  const int dwid = blockIdx.x * 256 + threadIdx.x;
  const int dw   = dwid & 7;
  const int lane = (dwid >> 3) & 31;
  const int fi   = dwid >> 8;
  const int kc   = fi % N_CHUNKS;
  const int ct   = fi / N_CHUNKS;
  const int hi   = lane >> 4;
  const int m    = lane & 15;
  const int cout = ct * 16 + m;
  const int g    = dw >> 1;
  const int o0   = (dw & 1) * 4;
  const int kb   = kc * 64 + (g >> 1) * 32 + hi * 16 + (g & 1) * 8 + o0;
  uint32_t v = 0;
  #pragma unroll
  for (int j = 0; j < 4; ++j) {
    const int k   = kb + j;
    const int tap = k >> 7;
    const int cin = k & 127;
    const uint8_t b = (uint8_t)w[((size_t)cout * C_IN + cin) * 9 + tap];
    v |= (uint32_t)b << (8 * j);
  }
  ((uint32_t*)wfrag)[dwid] = v;
}

__global__ void __launch_bounds__(128) weight_sum_kernel(
    const int8_t* __restrict__ w, int* __restrict__ wsum) {
  const int cout = blockIdx.x;
  const int c    = threadIdx.x;
  const int8_t* src = w + ((size_t)cout * C_IN + c) * 9;
  int s = 0;
  #pragma unroll
  for (int tap = 0; tap < 9; ++tap) s += (int)src[tap];
  __shared__ int red[128];
  red[c] = s;
  __syncthreads();
  #pragma unroll
  for (int off = 64; off > 0; off >>= 1) {
    if (c < off) red[c] += red[c + off];
    __syncthreads();
  }
  if (c == 0) wsum[cout] = red[0];
}

__device__ __forceinline__ v8i load_frag32(const int8_t* p) {
  v4i t0 = *(const v4i*)(p +  0);
  v4i t1 = *(const v4i*)(p + 16);
  v8i a;
  a[0] = t0[0]; a[1] = t0[1]; a[2] = t0[2]; a[3] = t0[3];
  a[4] = t1[0]; a[5] = t1[1]; a[6] = t1[2]; a[7] = t1[3];
  return a;
}
__device__ __forceinline__ v8i load_b_frag(const int8_t* p) {
  v4i t0 = *(const v4i*)(p +  0);
  v4i t1 = *(const v4i*)(p + 32);
  v8i b;
  b[0] = t0[0]; b[1] = t0[1]; b[2] = t0[2]; b[3] = t0[3];
  b[4] = t1[0]; b[5] = t1[1]; b[6] = t1[2]; b[7] = t1[3];
  return b;
}

__global__ void __launch_bounds__(128) conv_wmma_kernel(
    const int8_t* __restrict__ nhwc, const int8_t* __restrict__ wfrag,
    const int* __restrict__ wsum, const float* __restrict__ bias,
    float* __restrict__ out) {
  const int lane = threadIdx.x & 31;
  const int wave = threadIdx.x >> 5;
  const int hi   = lane >> 4;
  const int col  = lane & 15;

  const int ptile0 = (blockIdx.x * 4 + wave) * 2;
  int pn[2], poh[2], pow_[2];
  #pragma unroll
  for (int pt = 0; pt < 2; ++pt) {
    const int p   = (ptile0 + pt) * 16 + col;
    const int n   = p / (H_OUT * W_OUT);
    const int rem = p - n * (H_OUT * W_OUT);
    const int oh  = rem / W_OUT;
    pn[pt] = n; poh[pt] = oh; pow_[pt] = rem - oh * W_OUT;
  }
  const int cbase = blockIdx.y * 64;
  const int ct0   = blockIdx.y * 4;

  const v8i zero = {};
  v8i acc[4][2];
  v8i asum[2];
  #pragma unroll
  for (int q = 0; q < 4; ++q) { acc[q][0] = zero; acc[q][1] = zero; }
  asum[0] = zero; asum[1] = zero;
  v8i ones;
  #pragma unroll
  for (int i = 0; i < 8; ++i) ones[i] = 0x01010101;

  const int8_t* abase = wfrag + (size_t)lane * 32;

  #pragma unroll
  for (int t = 0; t < 9; ++t) {
    const int kh = t / 3, kw = t % 3;
    const int8_t* bptr[2];
    #pragma unroll
    for (int pt = 0; pt < 2; ++pt)
      bptr[pt] = nhwc +
          ((size_t)(((pn[pt] * H_IN) + poh[pt] + kh) * W_IN +
                    (pow_[pt] + kw)) << 7) + (hi << 4);
    #pragma unroll
    for (int h = 0; h < 2; ++h) {
      const int kc = t * 2 + h;
      v8i B0 = load_b_frag(bptr[0] + h * 64);
      v8i B1 = load_b_frag(bptr[1] + h * 64);
      asum[0] = __builtin_amdgcn_wmma_i32_16x16x64_iu8(
          true, ones, true, B0, asum[0], false, false);
      asum[1] = __builtin_amdgcn_wmma_i32_16x16x64_iu8(
          true, ones, true, B1, asum[1], false, false);
      #pragma unroll
      for (int q = 0; q < 4; ++q) {
        const v8i A = load_frag32(
            abase + (size_t)((ct0 + q) * N_CHUNKS + kc) * 1024);
        acc[q][0] = __builtin_amdgcn_wmma_i32_16x16x64_iu8(
            true, A, true, B0, acc[q][0], false, false);
        acc[q][1] = __builtin_amdgcn_wmma_i32_16x16x64_iu8(
            true, A, true, B1, acc[q][1], false, false);
      }
    }
  }

  const int as[2] = { asum[0][0], asum[1][0] };
  #pragma unroll
  for (int q = 0; q < 4; ++q) {
    #pragma unroll
    for (int r = 0; r < 8; ++r) {
      const int cout = cbase + q * 16 + r + hi * 8;
      const int wq   = wsum[cout];
      float bq = rintf(bias[cout] * 1e4f);
      bq = fminf(fmaxf(bq, -2147483648.0f), 2147483647.0f);
      const float bterm = bq * 1e-4f;
      #pragma unroll
      for (int pt = 0; pt < 2; ++pt) {
        const int S = acc[q][pt][r];
        const float val =
            1e-4f * (float)(S - 3 * as[pt] - 7 * wq + 21 * K_TOTAL) + bterm;
        out[(((size_t)pn[pt] * C_OUT + cout) * H_OUT + poh[pt]) * W_OUT +
            pow_[pt]] = val;
      }
    }
  }
}

extern "C" void kernel_launch(void* const* d_in, const int* in_sizes, int n_in,
                              void* d_out, int out_size, void* d_ws,
                              size_t ws_size, hipStream_t stream) {
  (void)in_sizes; (void)n_in; (void)out_size; (void)ws_size;
  const int8_t* in   = (const int8_t*)d_in[0];
  const int8_t* wt   = (const int8_t*)d_in[1];
  const float*  bias = (const float*)d_in[2];
  float* out = (float*)d_out;

  int8_t* nhwc  = (int8_t*)d_ws;
  int8_t* wfrag = nhwc + NHWC_BYTES;
  int*    wsum  = (int*)(wfrag + WFRAG_BYTES);

  repack_input_kernel<<<N_BATCH * H_IN * W_IN, 128, 0, stream>>>(in, nhwc);
  repack_weight_frag_kernel<<<288, 256, 0, stream>>>(wt, wfrag);
  weight_sum_kernel<<<C_OUT, 128, 0, stream>>>(wt, wsum);
  conv_wmma_kernel<<<dim3(PIX_TILES / 8, C_OUT / 64), 128, 0, stream>>>(
      nhwc, wfrag, wsum, bias, out);
}
